// Decoder_2121713844276
// MI455X (gfx1250) — matrix operand order repaired and checked by reading, not compiled, not run on hardware
//
#include <hip/hip_runtime.h>
#include <hip/hip_bf16.h>
#include <math.h>

typedef __attribute__((ext_vector_type(16))) __bf16       v16bf;
typedef __attribute__((ext_vector_type(8)))  float        v8f;
typedef __attribute__((ext_vector_type(8)))  unsigned int v8u;

__device__ __forceinline__ unsigned int pack_bf16x2(float a, float b) {
  unsigned short ua = __builtin_bit_cast(unsigned short, (__bf16)a);
  unsigned short ub = __builtin_bit_cast(unsigned short, (__bf16)b);
  return (unsigned int)ua | ((unsigned int)ub << 16);
}

// ---------------------------------------------------------------------------
// Implicit-GEMM conv via WMMA bf16 (fp32 accumulate), LDS-staged B tile.
//   out[b, oc, p] = bias[oc] + sum_k W[oc,k] * Im2col[k,p]
//   k = c*KH*KW + ky*KW + kx ; p = oy*W + ox
// WG = 128 threads (4 wave32). N tile = 32. B tile (32k x 32n bf16) staged
// cooperatively in LDS once per K-step and shared by all waves/M-tiles.
// MT = M-tiles per wave (1 or 2): wave w covers rows  base + w*16 (+ mt*64),
// so a WG covers M = 64*MT rows. MT=2 doubles WMMA issued per staged byte.
// Loader folds reflect/zero pad, x2 nearest upsample, channel concat.
// valmode: B read from prebuilt val[Ktot, HW] buffer (deformable conv GEMM).
// Requirements (all call sites satisfy): Ktot % 32 == 0, HW % 32 == 0.
// ---------------------------------------------------------------------------
struct ConvDesc {
  const unsigned short* w;   // bf16-packed (Cout, Ktot)
  const float* bias;         // (Cout) fp32
  const float* x0;           // first input tensor
  const float* x1;           // second (concat) input tensor, or nullptr
  float*       out;          // (B, Cout, H*W)
  int Cout, C0, Ktot;
  int H, W, KH, KW, pad, up, reflect, valmode;
  long x0Bs, x1Bs, outBs;    // per-batch element strides
};

template <int MT>
__global__ void conv_wmma_kernel(ConvDesc d) {
  __shared__ unsigned int smem[512];        // 32 pcol x 16 kpair dwords (2KB)
  const int tid   = threadIdx.x;
  const int lane  = tid & 31;
  const int wave  = tid >> 5;
  const int b     = blockIdx.z;
  const int mbase = blockIdx.y * (64 * MT) + wave * 16;
  const int nbase = blockIdx.x * 32;
  const int HW    = d.H * d.W;
  const int row   = lane & 15;
  const int half  = lane >> 4;

  const float* x0b = d.x0 + (size_t)b * d.x0Bs;
  const float* x1b = d.x1 ? d.x1 + (size_t)b * d.x1Bs : nullptr;

  // A-side: packed bf16 weight rows; fragment = 2 contiguous b128 dword loads
  bool aok[MT];
  const unsigned int* wrow32[MT];
#pragma unroll
  for (int mt = 0; mt < MT; ++mt) {
    int oc_a = mbase + mt * 64 + row;
    aok[mt]  = oc_a < d.Cout;
    wrow32[mt] = (const unsigned int*)d.w +
                 (size_t)(aok[mt] ? oc_a : 0) * (size_t)(d.Ktot >> 1);
  }

  // staging-thread constants (each thread owns 4 LDS dwords = 8 consecutive k)
  const int qbase  = tid * 4;
  const int s_pcol = qbase >> 4;            // column within N tile
  const int s_koff = (qbase & 15) * 2;      // k offset within K-step
  const int sp  = nbase + s_pcol;           // global output pixel
  const int soy = sp / d.W, sox = sp - soy * d.W;
  unsigned int* sdst = &smem[s_pcol * 16 + (qbase & 15)];

  const int upsh = (d.up == 2) ? 1 : 0;
  const int Ws   = d.W >> upsh;
  const int Hs   = d.H >> upsh;
  const int KHW  = d.KH * d.KW;

  v8f acc0[MT];
  v8f acc1[MT];
#pragma unroll
  for (int mt = 0; mt < MT; ++mt) { acc0[mt] = (v8f){}; acc1[mt] = (v8f){}; }

  for (int kb = 0; kb < d.Ktot; kb += 32) {
    // ---- gather 8 consecutive-k values for this thread's LDS slots ----
    float f[8];
    if (d.valmode) {
      const float* vp = x0b + (size_t)(kb + s_koff) * HW + sp;
#pragma unroll
      for (int t = 0; t < 8; ++t) f[t] = vp[(size_t)t * HW];
    } else {
      int k0 = kb + s_koff;                 // one decomposition per K-step
      int c  = k0 / KHW;
      int r  = k0 - c * KHW;
      int ky = r / d.KW;
      int kx = r - ky * d.KW;
#pragma unroll
      for (int t = 0; t < 8; ++t) {
        int iy = soy + ky - d.pad;
        int ix = sox + kx - d.pad;
        float v = 0.f;
        bool ok = true;
        if (d.reflect) {
          iy = iy < 0 ? -iy : (iy >= d.H ? 2 * d.H - 2 - iy : iy);
          ix = ix < 0 ? -ix : (ix >= d.W ? 2 * d.W - 2 - ix : ix);
        } else {
          ok = (iy >= 0) & (iy < d.H) & (ix >= 0) & (ix < d.W);
        }
        if (ok) {
          iy >>= upsh; ix >>= upsh;
          const float* src; int cc;
          if (c < d.C0) { src = x0b; cc = c; }
          else          { src = x1b; cc = c - d.C0; }
          v = src[((size_t)cc * Hs + iy) * Ws + ix];
        }
        f[t] = v;
        if (++kx == d.KW) { kx = 0; if (++ky == d.KH) { ky = 0; ++c; } }
      }
    }
    __syncthreads();                        // previous-step reads complete
    sdst[0] = pack_bf16x2(f[0], f[1]);
    sdst[1] = pack_bf16x2(f[2], f[3]);
    sdst[2] = pack_bf16x2(f[4], f[5]);
    sdst[3] = pack_bf16x2(f[6], f[7]);
    __syncthreads();                        // tile staged

    // ---- B fragments from LDS (contiguous dwords -> ds_load_b128) ----
    v8u bu0, bu1;
#pragma unroll
    for (int v = 0; v < 8; ++v) {
      bu0[v] = smem[(row)      * 16 + half * 8 + v];
      bu1[v] = smem[(row + 16) * 16 + half * 8 + v];
    }
    v16bf bf0 = __builtin_bit_cast(v16bf, bu0);
    v16bf bf1 = __builtin_bit_cast(v16bf, bu1);

    // ---- A fragments (contiguous k = 16*half + s, packed pairs, same order as B) ----
#pragma unroll
    for (int mt = 0; mt < MT; ++mt) {
      v8u aw;
      const unsigned int* wp = wrow32[mt] + (kb >> 1);
#pragma unroll
      for (int v = 0; v < 8; ++v)
        aw[v] = aok[mt] ? wp[half * 8 + v] : 0u;
      v16bf af = __builtin_bit_cast(v16bf, aw);

      acc0[mt] = __builtin_amdgcn_wmma_f32_16x16x32_bf16(
          false, af, false, bf0, (short)0, acc0[mt], false, false);
      acc1[mt] = __builtin_amdgcn_wmma_f32_16x16x32_bf16(
          false, af, false, bf1, (short)0, acc1[mt], false, false);
    }
  }

  // ---- store D (+bias): element r -> M = half*8 + r, N = lane&15 ----
  float* ob = d.out + (size_t)b * d.outBs;
  const int p0 = nbase + row;
  const int p1 = p0 + 16;
#pragma unroll
  for (int mt = 0; mt < MT; ++mt) {
#pragma unroll
    for (int r = 0; r < 8; ++r) {
      int oc = mbase + mt * 64 + half * 8 + r;
      if (oc >= d.Cout) continue;
      float bv = d.bias ? d.bias[oc] : 0.f;
      ob[(size_t)oc * HW + p0] = acc0[mt][r] + bv;
      ob[(size_t)oc * HW + p1] = acc1[mt][r] + bv;
    }
  }
}

// ---------------------------------------------------------------------------
// Weight pre-pack: fp32 -> bf16 bits
// ---------------------------------------------------------------------------
__global__ void pack_bf16_kernel(const float* __restrict__ src,
                                 unsigned short* __restrict__ dst, long n) {
  long i = (long)blockIdx.x * blockDim.x + threadIdx.x;
  if (i < n) dst[i] = __builtin_bit_cast(unsigned short, (__bf16)src[i]);
}

// ---------------------------------------------------------------------------
// Instance norm over HW per (b,c); optional relu; optional residual add.
// ---------------------------------------------------------------------------
__global__ void inorm_kernel(const float* __restrict__ src,
                             const float* __restrict__ resid,
                             float* __restrict__ dst, int HW, int relu) {
  __shared__ float ssum[256];
  __shared__ float ssq[256];
  const size_t base = (size_t)blockIdx.x * HW;
  float sum = 0.f, sq = 0.f;
  for (int i = threadIdx.x; i < HW; i += blockDim.x) {
    float v = src[base + i];
    sum += v; sq += v * v;
  }
  ssum[threadIdx.x] = sum; ssq[threadIdx.x] = sq;
  __syncthreads();
  for (int s = 128; s > 0; s >>= 1) {
    if ((int)threadIdx.x < s) {
      ssum[threadIdx.x] += ssum[threadIdx.x + s];
      ssq[threadIdx.x]  += ssq[threadIdx.x + s];
    }
    __syncthreads();
  }
  const float m   = ssum[0] / HW;
  const float var = ssq[0] / HW - m * m;
  const float inv = rsqrtf(var + 1e-5f);
  for (int i = threadIdx.x; i < HW; i += blockDim.x) {
    float y = (src[base + i] - m) * inv;
    if (relu) y = fmaxf(y, 0.f);
    if (resid) y += resid[base + i];
    dst[base + i] = y;
  }
}

// ---------------------------------------------------------------------------
// DCN val-tensor build (one batch): val[(c*9+k), p] = mask * bilinear(x)
// ---------------------------------------------------------------------------
__device__ __forceinline__ float dcn_tap(const float* xc, float yf, float xf,
                                         int H, int W) {
  bool valid = (yf >= 0.f) && (yf < (float)H) && (xf >= 0.f) && (xf < (float)W);
  int yc = (int)fminf(fmaxf(yf, 0.f), (float)(H - 1));
  int xi = (int)fminf(fmaxf(xf, 0.f), (float)(W - 1));
  float v = xc[(size_t)yc * W + xi];
  return valid ? v : 0.f;
}

__global__ void dcn_val_kernel(const float* __restrict__ x,    // (C,H,W)
                               const float* __restrict__ om,   // (27,H,W)
                               float* __restrict__ val,        // (C*9, HW)
                               int C, int H, int W) {
  const int HW = H * W;
  long idx = (long)blockIdx.x * blockDim.x + threadIdx.x;
  long total = (long)C * 9 * HW;
  if (idx >= total) return;
  int p = (int)(idx % HW);
  long t = idx / HW;
  int k = (int)(t % 9);
  int c = (int)(t / 9);
  int oy = p / W, ox = p - oy * W;
  int ky = k / 3, kx = k - ky * 3;
  float dyo = om[(size_t)(2 * k) * HW + p];
  float dxo = om[(size_t)(2 * k + 1) * HW + p];
  float msk = om[(size_t)(18 + k) * HW + p];
  msk = 1.f / (1.f + expf(-msk));
  float py = (float)(oy - 1 + ky) + dyo;
  float px = (float)(ox - 1 + kx) + dxo;
  float y0 = floorf(py), x0 = floorf(px);
  float dy = py - y0,    dx = px - x0;
  const float* xc = x + (size_t)c * HW;
  float acc = dcn_tap(xc, y0,       x0,       H, W) * (1.f - dy) * (1.f - dx)
            + dcn_tap(xc, y0,       x0 + 1.f, H, W) * (1.f - dy) * dx
            + dcn_tap(xc, y0 + 1.f, x0,       H, W) * dy * (1.f - dx)
            + dcn_tap(xc, y0 + 1.f, x0 + 1.f, H, W) * dy * dx;
  val[(size_t)(c * 9 + k) * HW + p] = acc * msk;
}

// ---------------------------------------------------------------------------
// Misc epilogues
// ---------------------------------------------------------------------------
__global__ void tanh_kernel(float* p, int n) {
  int i = blockIdx.x * blockDim.x + threadIdx.x;
  if (i < n) p[i] = tanhf(p[i]);
}

__global__ void init_scalar_kernel(float* p) {
  if (blockIdx.x == 0 && threadIdx.x == 0) *p = 0.f;
}

__global__ void absmean_kernel(const float* __restrict__ om, long total, int HW,
                               int Cfull, float scale, float* accum) {
  __shared__ float red[256];
  float s = 0.f;
  for (long i = (long)blockIdx.x * blockDim.x + threadIdx.x; i < total;
       i += (long)gridDim.x * blockDim.x) {
    int p = (int)(i % HW);
    long t = i / HW;
    int ch = (int)(t % 18);
    long b = t / 18;
    s += fabsf(om[((size_t)b * Cfull + ch) * (size_t)HW + p]);
  }
  red[threadIdx.x] = s;
  __syncthreads();
  for (int st = 128; st > 0; st >>= 1) {
    if ((int)threadIdx.x < st) red[threadIdx.x] += red[threadIdx.x + st];
    __syncthreads();
  }
  if (threadIdx.x == 0) atomicAdd(accum, red[0] * scale);
}

// ---------------------------------------------------------------------------
// Host orchestration
// ---------------------------------------------------------------------------
static void launch_conv(const ConvDesc& d, int B, hipStream_t s) {
  int HW = d.H * d.W;
  if (d.Cout >= 128) {
    dim3 grid(HW / 32, (d.Cout + 127) / 128, B);
    conv_wmma_kernel<2><<<grid, dim3(128), 0, s>>>(d);
  } else {
    dim3 grid(HW / 32, (d.Cout + 63) / 64, B);
    conv_wmma_kernel<1><<<grid, dim3(128), 0, s>>>(d);
  }
}

extern "C" void kernel_launch(void* const* d_in, const int* in_sizes, int n_in,
                              void* d_out, int out_size, void* d_ws, size_t ws_size,
                              hipStream_t stream) {
  const float* x      = (const float*)d_in[0];   // (4,256,32,32)
  const float* skip1  = (const float*)d_in[1];   // (4,64,128,128)
  const float* skip2  = (const float*)d_in[2];   // (4,128,64,64)
  const float* res_w  = (const float*)d_in[3];   // (2,2,256,256,3,3)
  const float* res_b  = (const float*)d_in[4];   // (2,2,256)
  const float* w1     = (const float*)d_in[5];   // (128,256,5,5)
  const float* b1     = (const float*)d_in[6];
  const float* w2     = (const float*)d_in[7];   // (64,256,5,5)
  const float* b2     = (const float*)d_in[8];
  const float* w3     = (const float*)d_in[9];   // (3,128,7,7)
  const float* b3     = (const float*)d_in[10];
  const float* off2_w = (const float*)d_in[11];  // (27,256,3,3)
  const float* off2_b = (const float*)d_in[12];
  const float* dcn2_w = (const float*)d_in[13];  // (128,128,3,3)
  const float* dcn2_b = (const float*)d_in[14];
  const float* off1_w = (const float*)d_in[15];  // (27,128,3,3)
  const float* off1_b = (const float*)d_in[16];
  const float* dcn1_w = (const float*)d_in[17];  // (64,64,3,3)
  const float* dcn1_b = (const float*)d_in[18];

  const int B = 4;
  float* out = (float*)d_out;

  // ---- workspace layout (fp32 elements) ----
  float* ws = (float*)d_ws;
  size_t woff = 0;
  auto walloc = [&](size_t n) {
    float* p = ws + woff;
    woff += (n + 255) & ~(size_t)255;
    return p;
  };
  float* craw  = walloc((size_t)B * 64 * 128 * 128);   // conv scratch (max raw out)
  float* bufA  = walloc((size_t)B * 256 * 32 * 32);    // resblock "out"
  float* bufB  = walloc((size_t)B * 256 * 32 * 32);    // resblock "h"
  float* t1    = walloc((size_t)B * 128 * 64 * 64);    // stage-1 features
  float* o2buf = walloc((size_t)B * 27 * 64 * 64);     // off2 conv out
  float* pre2  = walloc((size_t)B * 128 * 64 * 64);    // dcn2 out
  float* t2    = walloc((size_t)B * 64 * 128 * 128);   // stage-2 features
  float* o1buf = walloc((size_t)B * 27 * 128 * 128);   // off1 conv out
  float* pre1  = walloc((size_t)B * 64 * 128 * 128);   // dcn1 out
  float* val   = walloc((size_t)64 * 9 * 128 * 128);   // per-batch DCN val tensor
  unsigned short* warena = (unsigned short*)walloc(2200000);  // bf16 weight arena
  (void)ws_size; (void)in_sizes; (void)n_in; (void)out_size;

  // ---- pre-pack all conv weights to bf16 (A-side of the WMMA GEMMs) ----
  unsigned short* wptr = warena;
  auto pack = [&](const float* src, size_t n) {
    unsigned short* p = wptr;
    wptr += (n + 15) & ~(size_t)15;
    pack_bf16_kernel<<<dim3((unsigned)((n + 255) / 256)), dim3(256), 0, stream>>>(
        src, p, (long)n);
    return p;
  };
  unsigned short* wres_p  = pack(res_w,  (size_t)4 * 256 * 256 * 9);
  unsigned short* w1_p    = pack(w1,     (size_t)128 * 256 * 25);
  unsigned short* w2_p    = pack(w2,     (size_t)64 * 256 * 25);
  unsigned short* w3_p    = pack(w3,     (size_t)3 * 128 * 49);
  unsigned short* off2_p  = pack(off2_w, (size_t)27 * 256 * 9);
  unsigned short* dcn2_p  = pack(dcn2_w, (size_t)128 * 128 * 9);
  unsigned short* off1_p  = pack(off1_w, (size_t)27 * 128 * 9);
  unsigned short* dcn1_p  = pack(dcn1_w, (size_t)64 * 64 * 9);

  ConvDesc d{};

  // ======================= residual blocks (256ch, 32x32) ==================
  for (int i = 0; i < 2; ++i) {
    const float* cur = (i == 0) ? x : bufA;
    d = ConvDesc{};
    d.w = wres_p + (size_t)(i * 2 + 0) * 256 * 256 * 9;
    d.bias = res_b + (size_t)(i * 2 + 0) * 256;
    d.x0 = cur; d.x1 = nullptr; d.out = craw;
    d.Cout = 256; d.C0 = 256; d.Ktot = 256 * 9;
    d.H = 32; d.W = 32; d.KH = 3; d.KW = 3; d.pad = 1; d.up = 1;
    d.reflect = 1; d.valmode = 0;
    d.x0Bs = 256 * 1024; d.x1Bs = 0; d.outBs = 256 * 1024;
    launch_conv(d, B, stream);
    inorm_kernel<<<dim3(B * 256), dim3(256), 0, stream>>>(craw, nullptr, bufB, 1024, 1);
    d.w = wres_p + (size_t)(i * 2 + 1) * 256 * 256 * 9;
    d.bias = res_b + (size_t)(i * 2 + 1) * 256;
    d.x0 = bufB;
    launch_conv(d, B, stream);
    inorm_kernel<<<dim3(B * 256), dim3(256), 0, stream>>>(craw, cur, bufA, 1024, 0);
  }

  // ================= stage 1: up2 + refpad2 + 5x5 (256->128 @64x64) ========
  d = ConvDesc{};
  d.w = w1_p; d.bias = b1; d.x0 = bufA; d.x1 = nullptr; d.out = craw;
  d.Cout = 128; d.C0 = 256; d.Ktot = 256 * 25;
  d.H = 64; d.W = 64; d.KH = 5; d.KW = 5; d.pad = 2; d.up = 2;
  d.reflect = 1; d.valmode = 0;
  d.x0Bs = 256 * 1024; d.x1Bs = 0; d.outBs = 128 * 4096;
  launch_conv(d, B, stream);
  inorm_kernel<<<dim3(B * 128), dim3(256), 0, stream>>>(craw, nullptr, t1, 4096, 1);

  // ---- off2 conv: concat(t1, skip2) zero-pad 3x3 -> 27ch ----
  d = ConvDesc{};
  d.w = off2_p; d.bias = off2_b; d.x0 = t1; d.x1 = skip2; d.out = o2buf;
  d.Cout = 27; d.C0 = 128; d.Ktot = 256 * 9;
  d.H = 64; d.W = 64; d.KH = 3; d.KW = 3; d.pad = 1; d.up = 1;
  d.reflect = 0; d.valmode = 0;
  d.x0Bs = 128 * 4096; d.x1Bs = 128 * 4096; d.outBs = 27 * 4096;
  launch_conv(d, B, stream);

  // ---- dcn2: per batch, build val then WMMA GEMM (Ktot=1152, 128->128) ----
  for (int b = 0; b < B; ++b) {
    long nval = (long)128 * 9 * 4096;
    dcn_val_kernel<<<dim3((unsigned)((nval + 255) / 256)), dim3(256), 0, stream>>>(
        skip2 + (size_t)b * 128 * 4096, o2buf + (size_t)b * 27 * 4096, val,
        128, 64, 64);
    d = ConvDesc{};
    d.w = dcn2_p; d.bias = dcn2_b; d.x0 = val; d.x1 = nullptr;
    d.out = pre2 + (size_t)b * 128 * 4096;
    d.Cout = 128; d.C0 = 128; d.Ktot = 128 * 9;
    d.H = 64; d.W = 64; d.KH = 3; d.KW = 3; d.pad = 1; d.up = 1;
    d.reflect = 0; d.valmode = 1;
    d.x0Bs = 0; d.x1Bs = 0; d.outBs = 0;
    launch_conv(d, 1, stream);
  }

  // ============ stage 2: up2 + refpad2 + 5x5 (concat(pre2,t1) -> 64) =======
  d = ConvDesc{};
  d.w = w2_p; d.bias = b2; d.x0 = pre2; d.x1 = t1; d.out = craw;
  d.Cout = 64; d.C0 = 128; d.Ktot = 256 * 25;
  d.H = 128; d.W = 128; d.KH = 5; d.KW = 5; d.pad = 2; d.up = 2;
  d.reflect = 1; d.valmode = 0;
  d.x0Bs = 128 * 4096; d.x1Bs = 128 * 4096; d.outBs = 64 * 16384;
  launch_conv(d, B, stream);
  inorm_kernel<<<dim3(B * 64), dim3(256), 0, stream>>>(craw, nullptr, t2, 16384, 1);

  // ---- off1 conv: concat(t2, skip1) zero-pad 3x3 -> 27ch ----
  d = ConvDesc{};
  d.w = off1_p; d.bias = off1_b; d.x0 = t2; d.x1 = skip1; d.out = o1buf;
  d.Cout = 27; d.C0 = 64; d.Ktot = 128 * 9;
  d.H = 128; d.W = 128; d.KH = 3; d.KW = 3; d.pad = 1; d.up = 1;
  d.reflect = 0; d.valmode = 0;
  d.x0Bs = 64 * 16384; d.x1Bs = 64 * 16384; d.outBs = 27 * 16384;
  launch_conv(d, B, stream);

  // ---- dcn1: per batch (Ktot=576, 64->64 @128x128) ----
  for (int b = 0; b < B; ++b) {
    long nval = (long)64 * 9 * 16384;
    dcn_val_kernel<<<dim3((unsigned)((nval + 255) / 256)), dim3(256), 0, stream>>>(
        skip1 + (size_t)b * 64 * 16384, o1buf + (size_t)b * 27 * 16384, val,
        64, 128, 128);
    d = ConvDesc{};
    d.w = dcn1_p; d.bias = dcn1_b; d.x0 = val; d.x1 = nullptr;
    d.out = pre1 + (size_t)b * 64 * 16384;
    d.Cout = 64; d.C0 = 64; d.Ktot = 64 * 9;
    d.H = 128; d.W = 128; d.KH = 3; d.KW = 3; d.pad = 1; d.up = 1;
    d.reflect = 0; d.valmode = 1;
    d.x0Bs = 0; d.x1Bs = 0; d.outBs = 0;
    launch_conv(d, 1, stream);
  }

  // ======== final: refpad3 + 7x7 (concat(pre1,t2) 128 -> 3) + tanh =========
  d = ConvDesc{};
  d.w = w3_p; d.bias = b3; d.x0 = pre1; d.x1 = t2; d.out = out;
  d.Cout = 3; d.C0 = 64; d.Ktot = 128 * 49;
  d.H = 128; d.W = 128; d.KH = 7; d.KW = 7; d.pad = 3; d.up = 1;
  d.reflect = 1; d.valmode = 0;
  d.x0Bs = 64 * 16384; d.x1Bs = 64 * 16384; d.outBs = 3 * 16384;
  launch_conv(d, B, stream);

  const int nout = B * 3 * 16384;  // 196608
  tanh_kernel<<<dim3((nout + 255) / 256), dim3(256), 0, stream>>>(out, nout);

  // ======== offset_sum = 0.5*(mean|off1| + mean|off2|) -> out[196608] ======
  float* accum = out + nout;
  init_scalar_kernel<<<dim3(1), dim3(32), 0, stream>>>(accum);
  {
    long tot1 = (long)B * 18 * 16384;
    absmean_kernel<<<dim3(256), dim3(256), 0, stream>>>(
        o1buf, tot1, 16384, 27, 0.5f / (float)tot1, accum);
    long tot2 = (long)B * 18 * 4096;
    absmean_kernel<<<dim3(256), dim3(256), 0, stream>>>(
        o2buf, tot2, 4096, 27, 0.5f / (float)tot2, accum);
  }
}
